// FourOverSixGptOssMLP_50354196579052
// MI455X (gfx1250) — matrix operand order repaired and checked by reading, not compiled, not run on hardware
//
#include <hip/hip_runtime.h>
#include <hip/hip_bf16.h>
#include <stdint.h>

// ---------------------------------------------------------------------------
// GPT-OSS MoE MLP for MI455X (gfx1250), wave32 + v_wmma_f32_16x16x32_bf16,
// with Tensor Data Mover (tensor_load_to_lds) staging for the down-proj A tile.
// Top-2 sparse expert execution (51.5 GFLOP vs 206 GFLOP dense reference).
// ---------------------------------------------------------------------------

#define T_TOK   4096
#define HDIM    1024
#define IDIM    1024
#define NEXP    8
#define LDSW    18        // padded LDS row stride in dwords (16 data + 2 pad)
#define ACT_ROWS (2 * T_TOK + 128)   // 2T assignments + tile slack

typedef __attribute__((ext_vector_type(16))) __bf16   v16bf;
typedef __attribute__((ext_vector_type(8)))  float    v8f;
typedef __attribute__((ext_vector_type(8)))  uint32_t v8u;
typedef __attribute__((ext_vector_type(4)))  uint32_t tdm_g0_t;
typedef __attribute__((ext_vector_type(8)))  int      tdm_g1_t;
typedef __attribute__((ext_vector_type(4)))  int      tdm_g2_t;

// ---- fp32 -> bf16 (round-to-nearest-even) packing helpers -----------------
__device__ __forceinline__ uint32_t pack_bf16_2(float a, float b) {
    uint32_t ua = __float_as_uint(a), ub = __float_as_uint(b);
    ua = (ua + 0x7FFFu + ((ua >> 16) & 1u)) >> 16;
    ub = (ub + 0x7FFFu + ((ub >> 16) & 1u)) >> 16;
    return (ua & 0xFFFFu) | (ub << 16);
}
__device__ __forceinline__ uint16_t bf16_1(float a) {
    uint32_t u = __float_as_uint(a);
    return (uint16_t)((u + 0x7FFFu + ((u >> 16) & 1u)) >> 16);
}

// ---- WMMA fragment loaders (per-lane, from padded LDS) --------------------
// A 16x32 bf16: lanes 0-15 hold K0..15 contiguous, lanes 16-31 K16..31
__device__ __forceinline__ v16bf frag_a(const uint32_t* p /* row*LDSW + kb2 */) {
    v8u u;
#pragma unroll
    for (int j = 0; j < 8; ++j) u[j] = p[j];
    return __builtin_bit_cast(v16bf, u);
}
// B 32x16 bf16: lanes 0-15 hold K0..15 contiguous, lanes 16-31 K16..31
__device__ __forceinline__ v16bf frag_b(const uint32_t* p /* n*LDSW + kb8 */) {
    v8u u;
#pragma unroll
    for (int j = 0; j < 8; ++j) u[j] = p[j];
    return __builtin_bit_cast(v16bf, u);
}
__device__ __forceinline__ v8f wmma_bf16(v16bf a, v16bf b, v8f c) {
    return __builtin_amdgcn_wmma_f32_16x16x32_bf16(false, a, false, b, (short)0, c,
                                                   false, false);
}

// ---- Tensor Data Mover: 2-D b32 tile load with hardware LDS padding -------
// Copies tile_h rows x 16 dwords from global (row pitch pitch_dw dwords) into
// LDS at lds_byte_off, inserting 2 pad dwords after every 16 data dwords
// (pad_interval=3 -> 16 dwords, pad_amount=1 -> 2 dwords) => LDSW=18 layout.
__device__ __forceinline__ void tdm_load_tile(uint32_t lds_byte_off,
                                              const uint32_t* gsrc,
                                              uint32_t tile_w_dw,
                                              uint32_t tile_h,
                                              uint32_t pitch_dw) {
    uint64_t ga = (uint64_t)(uintptr_t)gsrc;
    tdm_g0_t g0 = (tdm_g0_t)0u;
    g0[0] = 1u;                                   // count=1 (valid user D#)
    g0[1] = lds_byte_off;                         // lds_addr (bytes)
    g0[2] = (uint32_t)ga;                         // global_addr[31:0]
    g0[3] = (uint32_t)((ga >> 32) & 0x01FFFFFFu)  // global_addr[56:32]
          | (2u << 30);                           // type=2 ("image")
    tdm_g1_t g1 = (tdm_g1_t)0;
    g1[0] = (int)((2u << 16)                      // data_size = 4 bytes
                | (1u << 20)                      // pad_enable
                | (3u << 22)                      // pad_interval: 16 dwords
                | (1u << 25));                    // pad_amount:   2 dwords
    g1[1] = (int)((tile_w_dw & 0xFFFFu) << 16);   // tensor_dim0[15:0]
    g1[2] = (int)(((tile_w_dw >> 16) & 0xFFFFu)   // tensor_dim0[31:16]
                | ((tile_h & 0xFFFFu) << 16));    // tensor_dim1[15:0]
    g1[3] = (int)(((tile_h >> 16) & 0xFFFFu)      // tensor_dim1[31:16]
                | ((tile_w_dw & 0xFFFFu) << 16)); // tile_dim0
    g1[4] = (int)(tile_h & 0xFFFFu);              // tile_dim1 (tile_dim2=0)
    g1[5] = (int)pitch_dw;                        // tensor_dim0_stride[31:0]
    // g1[6], g1[7] = stride hi / dim1_stride = 0 (2-D tile)
    tdm_g2_t gz = (tdm_g2_t)0;
    tdm_g1_t gz8 = (tdm_g1_t)0;
    // clang-23 / therock-10.0 form: 6 args (g0, g1, g2, g3, g4, cpol)
    __builtin_amdgcn_tensor_load_to_lds(g0, g1, gz, gz, gz8, 0);
}

// ---------------------------------------------------------------------------
// Kernel 0: zero output accumulator (T*H floats) and expert counters
// ---------------------------------------------------------------------------
__global__ void zero_kernel(float* __restrict__ out, uint32_t* __restrict__ cnt) {
    size_t i = ((size_t)blockIdx.x * blockDim.x + threadIdx.x) * 4;
    float4 z = {0.f, 0.f, 0.f, 0.f};
    *(float4*)(out + i) = z;
    if (blockIdx.x == 0 && threadIdx.x < NEXP) cnt[threadIdx.x] = 0u;
}

// ---------------------------------------------------------------------------
// Kernel 1: router — logits, top-2, softmax, build per-expert token lists
// ---------------------------------------------------------------------------
__global__ __launch_bounds__(256) void router_kernel(
    const float* __restrict__ x, const float* __restrict__ rw,
    const float* __restrict__ rb, float* __restrict__ scores_out,
    uint32_t* __restrict__ cnt, uint32_t* __restrict__ tok,
    float* __restrict__ wts) {
    __shared__ float lw[NEXP][HDIM];            // 32 KB router weights
    for (int i = threadIdx.x; i < NEXP * HDIM; i += 256)
        lw[i >> 10][i & (HDIM - 1)] = rw[i];
    __syncthreads();

    int t = blockIdx.x * 256 + threadIdx.x;
    const float* xr = x + (size_t)t * HDIM;
    float acc[NEXP];
#pragma unroll
    for (int e = 0; e < NEXP; ++e) acc[e] = rb[e];
    for (int h = 0; h < HDIM; ++h) {
        float xv = xr[h];
#pragma unroll
        for (int e = 0; e < NEXP; ++e) acc[e] = fmaf(xv, lw[e][h], acc[e]);
    }
    // top-2 (lowest index wins ties, like lax.top_k)
    int i0 = 0;
#pragma unroll
    for (int e = 1; e < NEXP; ++e) if (acc[e] > acc[i0]) i0 = e;
    int i1 = (i0 == 0) ? 1 : 0;
#pragma unroll
    for (int e = 0; e < NEXP; ++e) if (e != i0 && acc[e] > acc[i1]) i1 = e;
    float e1 = __expf(acc[i1] - acc[i0]);       // v0 is max -> e0 = 1
    float inv = 1.0f / (1.0f + e1);
    float w0 = inv, w1 = e1 * inv;
    scores_out[t * 2 + 0] = w0;
    scores_out[t * 2 + 1] = w1;
    uint32_t p0 = atomicAdd(&cnt[i0], 1u);
    tok[i0 * T_TOK + p0] = (uint32_t)t; wts[i0 * T_TOK + p0] = w0;
    uint32_t p1 = atomicAdd(&cnt[i1], 1u);
    tok[i1 * T_TOK + p1] = (uint32_t)t; wts[i1 * T_TOK + p1] = w1;
}

// ---------------------------------------------------------------------------
// Kernel 2: 8-entry exclusive scan -> activation row bases
// ---------------------------------------------------------------------------
__global__ void scan_kernel(const uint32_t* __restrict__ cnt,
                            uint32_t* __restrict__ base) {
    if (threadIdx.x == 0) {
        uint32_t s = 0;
#pragma unroll
        for (int e = 0; e < NEXP; ++e) { base[e] = s; s += cnt[e]; }
    }
}

// ---------------------------------------------------------------------------
// Kernel 3: gate_up GEMM + GLU. Block tile: 128 tokens x 64 I-channels.
// 8 waves (4x2); each wave: 32x32 of gate AND up (8 WMMAs / K-step).
// ---------------------------------------------------------------------------
__global__ __launch_bounds__(256) void gateup_kernel(
    const float* __restrict__ x, const float* __restrict__ gup,
    const float* __restrict__ gub, const uint32_t* __restrict__ cnt,
    const uint32_t* __restrict__ base, const uint32_t* __restrict__ tok,
    uint16_t* __restrict__ act) {
    const int e = blockIdx.z;
    const uint32_t n_e = cnt[e];
    const uint32_t row0 = blockIdx.x * 128;
    if (row0 >= n_e) return;                      // uniform early exit
    const int i0 = blockIdx.y * 64;               // I-channel tile base

    __shared__ uint32_t sA[128 * LDSW];           // tokens x K  (bf16 pairs)
    __shared__ uint32_t sBg[64 * LDSW];           // gate cols x K
    __shared__ uint32_t sBu[64 * LDSW];           // up   cols x K
    __shared__ uint32_t sTok[128];

    const int tid = threadIdx.x;
    if (tid < 128) {
        uint32_t p = row0 + (uint32_t)tid;
        sTok[tid] = tok[e * T_TOK + ((p < n_e) ? p : 0u)];
    }
    __syncthreads();

    const int lane = tid & 31, wid = tid >> 5;
    const int wr = wid >> 1, wc = wid & 1;        // 4x2 wave grid
    const int nl = lane & 15;
    const int kb2 = (lane < 16) ? 0 : 8;          // A frag word offset
    const int kb8 = (lane < 16) ? 0 : 8;          // B frag word offset
    const int hb  = (lane < 16) ? 0 : 8;          // C row-half offset

    v8f accG[2][2], accU[2][2];
#pragma unroll
    for (int a = 0; a < 2; ++a)
#pragma unroll
        for (int b = 0; b < 2; ++b) {
            accG[a][b] = (v8f){0, 0, 0, 0, 0, 0, 0, 0};
            accU[a][b] = (v8f){0, 0, 0, 0, 0, 0, 0, 0};
        }

    const float* W = gup + (size_t)e * HDIM * (2 * IDIM);

    for (int k0 = 0; k0 < HDIM; k0 += 32) {
        // stage A: 128 rows x 16 kpairs (fp32 -> bf16)
#pragma unroll
        for (int it = 0; it < 8; ++it) {
            int idx = tid + it * 256;
            int m = idx >> 4, kp = idx & 15;
            const float* src = x + (size_t)sTok[m] * HDIM + k0 + kp * 2;
            sA[m * LDSW + kp] = pack_bf16_2(src[0], src[1]);
        }
        // stage Bg/Bu: deinterleave gate/up while transposing into [n][k]
#pragma unroll
        for (int it = 0; it < 4; ++it) {
            int idx = tid + it * 256;
            int n = idx >> 4, kp = idx & 15;
            const float* s0 = W + (size_t)(k0 + kp * 2) * (2 * IDIM) + 2 * (i0 + n);
            const float* s1 = s0 + (2 * IDIM);
            __builtin_prefetch(s0 + 64 * (2 * IDIM), 0, 1);   // K tile after next (64 rows ahead)
            sBg[n * LDSW + kp] = pack_bf16_2(s0[0], s1[0]);
            sBu[n * LDSW + kp] = pack_bf16_2(s0[1], s1[1]);
        }
        __syncthreads();

        v16bf fa[2];
#pragma unroll
        for (int tm = 0; tm < 2; ++tm)
            fa[tm] = frag_a(&sA[(wr * 32 + tm * 16 + nl) * LDSW + kb2]);
#pragma unroll
        for (int tn = 0; tn < 2; ++tn) {
            v16bf bg = frag_b(&sBg[(wc * 32 + tn * 16 + nl) * LDSW + kb8]);
            v16bf bu = frag_b(&sBu[(wc * 32 + tn * 16 + nl) * LDSW + kb8]);
#pragma unroll
            for (int tm = 0; tm < 2; ++tm) {
                accG[tm][tn] = wmma_bf16(fa[tm], bg, accG[tm][tn]);
                accU[tm][tn] = wmma_bf16(fa[tm], bu, accU[tm][tn]);
            }
        }
        __syncthreads();
    }

    // GLU epilogue -> bf16 activations in workspace
    const float* bias = gub + (size_t)e * (2 * IDIM);
    const size_t slot0 = (size_t)base[e] + row0;
#pragma unroll
    for (int tn = 0; tn < 2; ++tn) {
        int iCh = i0 + wc * 32 + tn * 16 + nl;
        float bg = bias[2 * iCh], bu = bias[2 * iCh + 1];
#pragma unroll
        for (int tm = 0; tm < 2; ++tm) {
#pragma unroll
            for (int v = 0; v < 8; ++v) {
                int r = wr * 32 + tm * 16 + hb + v;
                float g = accG[tm][tn][v] + bg;
                float u = accU[tm][tn][v] + bu;
                g = fminf(g, 7.0f);
                u = fminf(fmaxf(u, -7.0f), 7.0f);
                float glu = g / (1.0f + __expf(-1.702f * g));
                float a = (u + 1.0f) * glu;
                act[(slot0 + r) * IDIM + iCh] = bf16_1(a);
            }
        }
    }
}

// ---------------------------------------------------------------------------
// Kernel 4: down GEMM + weighted combine. Block tile: 128 tokens x 128 H.
// A tile (already bf16) staged by the Tensor Data Mover with hardware LDS
// padding reproducing the LDSW=18 layout; B staged with VALU bf16 convert.
// 8 waves (4x2); each wave 32x64 -> 8 WMMAs / K-step.
// ---------------------------------------------------------------------------
__global__ __launch_bounds__(256) void down_kernel(
    const uint16_t* __restrict__ act, const float* __restrict__ dw,
    const float* __restrict__ db, const uint32_t* __restrict__ cnt,
    const uint32_t* __restrict__ base, const uint32_t* __restrict__ tok,
    const float* __restrict__ wts, float* __restrict__ out) {
    const int e = blockIdx.z;
    const uint32_t n_e = cnt[e];
    const uint32_t row0 = blockIdx.x * 128;
    if (row0 >= n_e) return;
    const int h0 = blockIdx.y * 128;

    __shared__ uint32_t sA[128 * LDSW];
    __shared__ uint32_t sB[128 * LDSW];

    const int tid = threadIdx.x;
    const int lane = tid & 31, wid = tid >> 5;
    const int wr = wid >> 1, wc = wid & 1;
    const int nl = lane & 15;
    const int kb2 = (lane < 16) ? 0 : 8;
    const int kb8 = (lane < 16) ? 0 : 8;
    const int hb  = (lane < 16) ? 0 : 8;

    v8f acc[2][4];
#pragma unroll
    for (int a = 0; a < 2; ++a)
#pragma unroll
        for (int b = 0; b < 4; ++b) acc[a][b] = (v8f){0, 0, 0, 0, 0, 0, 0, 0};

    const float* W = dw + (size_t)e * IDIM * HDIM;
    const uint32_t* actU = (const uint32_t*)act;        // bf16 pairs
    const size_t aBase = ((size_t)base[e] + row0) * (IDIM / 2);
    const uint32_t sA_off = (uint32_t)(uintptr_t)(void*)sA;  // LDS byte offset

    for (int k0 = 0; k0 < IDIM; k0 += 32) {
        // stage A via TDM: 128 rows x 16 dwords, row pitch IDIM/2 dwords,
        // hardware-padded to the LDSW=18 swizzle layout. One wave issues.
        if (wid == 0) {
            tdm_load_tile(sA_off, actU + aBase + (size_t)(k0 >> 1),
                          /*tile_w_dw=*/16, /*tile_h=*/128,
                          /*pitch_dw=*/IDIM / 2);
        }
        // stage B: transpose [k][h] -> [n][k] with bf16 conversion (overlaps TDM)
#pragma unroll
        for (int it = 0; it < 8; ++it) {
            int idx = tid + it * 256;
            int n = idx >> 4, kp = idx & 15;
            const float* s0 = W + (size_t)(k0 + kp * 2) * HDIM + h0 + n;
            __builtin_prefetch(s0 + 32 * HDIM, 0, 1);   // next K tile
            sB[n * LDSW + kp] = pack_bf16_2(s0[0], s0[HDIM]);
        }
        if (wid == 0) __builtin_amdgcn_s_wait_tensorcnt(0);
        __syncthreads();

        v16bf fa[2];
#pragma unroll
        for (int tm = 0; tm < 2; ++tm)
            fa[tm] = frag_a(&sA[(wr * 32 + tm * 16 + nl) * LDSW + kb2]);
#pragma unroll
        for (int tn = 0; tn < 4; ++tn) {
            v16bf fb = frag_b(&sB[(wc * 64 + tn * 16 + nl) * LDSW + kb8]);
#pragma unroll
            for (int tm = 0; tm < 2; ++tm)
                acc[tm][tn] = wmma_bf16(fa[tm], fb, acc[tm][tn]);
        }
        __syncthreads();
    }

    // epilogue: out[t, h] += comb * (acc + down_bias[h]), guarded by count
#pragma unroll
    for (int tm = 0; tm < 2; ++tm) {
#pragma unroll
        for (int v = 0; v < 8; ++v) {
            uint32_t pos = row0 + (uint32_t)(wr * 32 + tm * 16 + hb + v);
            if (pos >= n_e) continue;
            uint32_t t = tok[e * T_TOK + pos];
            float wcomb = wts[e * T_TOK + pos];
            float* orow = out + (size_t)t * HDIM;
#pragma unroll
            for (int tn = 0; tn < 4; ++tn) {
                int h = h0 + wc * 64 + tn * 16 + nl;
                float valv = wcomb * (acc[tm][tn][v] + db[e * HDIM + h]);
                atomicAdd(&orow[h], valv);
            }
        }
    }
}

// ---------------------------------------------------------------------------
// Host-side launcher
// ---------------------------------------------------------------------------
extern "C" void kernel_launch(void* const* d_in, const int* in_sizes, int n_in,
                              void* d_out, int out_size, void* d_ws, size_t ws_size,
                              hipStream_t stream) {
    const float* x   = (const float*)d_in[0];   // [T, H]
    const float* rw  = (const float*)d_in[1];   // [E, H]
    const float* rb  = (const float*)d_in[2];   // [E]
    const float* gup = (const float*)d_in[3];   // [E, H, 2I]
    const float* gub = (const float*)d_in[4];   // [E, 2I]
    const float* dwn = (const float*)d_in[5];   // [E, I, H]
    const float* dbn = (const float*)d_in[6];   // [E, H]
    float* out = (float*)d_out;                 // [T*H] states ++ [T*2] scores

    // workspace layout (dword-indexed from d_ws)
    uint32_t* cnt  = (uint32_t*)d_ws;                         // 8
    uint32_t* bas  = cnt + 8;                                 // 8
    uint32_t* tok  = cnt + 16;                                // 8*T
    float*    wts  = (float*)(cnt + 16 + NEXP * T_TOK);       // 8*T
    uint16_t* act  = (uint16_t*)(cnt + 16 + 2 * NEXP * T_TOK);// ACT_ROWS*I bf16

    zero_kernel<<<(T_TOK * HDIM) / (256 * 4), 256, 0, stream>>>(out, cnt);
    router_kernel<<<T_TOK / 256, 256, 0, stream>>>(
        x, rw, rb, out + (size_t)T_TOK * HDIM, cnt, tok, wts);
    scan_kernel<<<1, 32, 0, stream>>>(cnt, bas);
    gateup_kernel<<<dim3(T_TOK / 128, IDIM / 64, NEXP), 256, 0, stream>>>(
        x, gup, gub, cnt, bas, tok, act);
    down_kernel<<<dim3(T_TOK / 128, HDIM / 128, NEXP), 256, 0, stream>>>(
        act, dwn, dbn, cnt, bas, tok, wts, out);
    (void)in_sizes; (void)n_in; (void)out_size; (void)ws_size;
}
